// Model_13761075216878
// MI455X (gfx1250) — compile-verified, matrix operand order checked by reading, not run on hardware
//
#include <hip/hip_runtime.h>
#include <cstdint>
#include <cstddef>

// ---------------------------------------------------------------------------
// xLSTM forward for MI455X (gfx1250, wave32, WMMA).
// All large matmuls use v_wmma_f32_16x16x32_bf16 (bf16 in, f32 accum).
// sLSTM recurrence: persistent kernel, 1 workgroup/head, h kept in LDS,
// R kept hot in L2 (512KB << 192MB), WMMA per step.
// Attention: flash-style online-rescaled accumulation, wave-per-tile.
// ---------------------------------------------------------------------------

typedef __attribute__((ext_vector_type(16))) __bf16 v16bf;
typedef __attribute__((ext_vector_type(8)))  float  v8f;

#define WMMA_BF16(a, b, c) \
  __builtin_amdgcn_wmma_f32_16x16x32_bf16(false, (a), false, (b), (short)0, (c), false, false)

static constexpr int Bb    = 8;
static constexpr int Ss    = 1024;
static constexpr int CIN   = 64;
static constexpr int Dd    = 512;
static constexpr int INNER = 1024;
static constexpr int ROWS  = Bb * Ss;   // 8192
static constexpr int FFUP  = 704;

// ---------------- bf16 helpers (storage-only, no __bf16 arithmetic) --------
__device__ __forceinline__ __bf16 f2bf(float f) {
  union { float f; unsigned u; } a; a.f = f;
  unsigned r = a.u + 0x7fffu + ((a.u >> 16) & 1u);
  unsigned short hv = (unsigned short)(r >> 16);
  __bf16 out; __builtin_memcpy(&out, &hv, 2); return out;
}
__device__ __forceinline__ float bf2f(__bf16 b) {
  unsigned short hv; __builtin_memcpy(&hv, &b, 2);
  union { unsigned u; float f; } a; a.u = ((unsigned)hv) << 16; return a.f;
}

// A/B fragment loader: dim0 indexed by (lane&15) with 'stride' elements per
// row, K pairs contiguous along dim1.  Matches the documented 16-bit 16x32
// A-layout (lanes 0-15: K {0..7,16..23}; lanes 16-31: K {8..15,24..31}).
__device__ __forceinline__ v16bf frag_load_pairs(const __bf16* __restrict__ base,
                                                 size_t stride, int lane) {
  int row = lane & 15, half = lane >> 4;
  union { v16bf v; unsigned u[8]; } r;
  const __bf16* p = base + (size_t)row * stride + half * 8;
#pragma unroll
  for (int i = 0; i < 8; ++i) {
    int k = (i < 4) ? (2 * i) : (16 + 2 * (i - 4));
    r.u[i] = *(const unsigned*)(p + k);
  }
  return r.v;
}

// B fragment from a K-major [k][n] matrix (scalar gathers), zeroing K>=kvalid.
__device__ __forceinline__ v16bf frag_load_kmajor(const __bf16* __restrict__ base,
                                                  size_t ldk, int lane, int kvalid) {
  int col = lane & 15, half = lane >> 4;
  union { v16bf v; unsigned short e[16]; } r;
#pragma unroll
  for (int i = 0; i < 8; ++i) {
    int k0 = ((i < 4) ? (2 * i) : (16 + 2 * (i - 4))) + half * 8;
#pragma unroll
    for (int j = 0; j < 2; ++j) {
      int k = k0 + j;
      unsigned short val = 0;
      if (k < kvalid) { __bf16 t = base[(size_t)k * ldk + col]; __builtin_memcpy(&val, &t, 2); }
      r.e[i * 2 + j] = val;
    }
  }
  return r.v;
}

__device__ __forceinline__ float block_reduce_256(float v, float* sbuf) {
  int lane = threadIdx.x & 31, wv = threadIdx.x >> 5;
#pragma unroll
  for (int m = 16; m >= 1; m >>= 1) v += __shfl_xor(v, m, 32);
  if (lane == 0) sbuf[wv] = v;
  __syncthreads();
  if (threadIdx.x == 0) {
    float r = 0.f;
    for (int i = 0; i < 8; ++i) r += sbuf[i];
    sbuf[0] = r;
  }
  __syncthreads();
  float r = sbuf[0];
  __syncthreads();
  return r;
}

// ---------------- weight prep ----------------------------------------------
__global__ void cvt_f32_bf16_kernel(const float* __restrict__ src,
                                    __bf16* __restrict__ dst, int n) {
  for (int i = blockIdx.x * blockDim.x + threadIdx.x; i < n; i += gridDim.x * blockDim.x)
    dst[i] = f2bf(src[i]);
}

// Wg (4 gates, 4 heads, 128, 128) -> WGP[h][k][g*128+e]  (K-major per head)
__global__ void pack_wg_kernel(const float* __restrict__ Wg, __bf16* __restrict__ dst) {
  int n = 4 * 128 * 512;
  for (int i = blockIdx.x * blockDim.x + threadIdx.x; i < n; i += gridDim.x * blockDim.x) {
    int h = i / (128 * 512); int rem = i % (128 * 512);
    int k = rem / 512, nn = rem % 512;
    int g = nn >> 7, e = nn & 127;
    dst[i] = f2bf(Wg[(((size_t)g * 4 + h) * 128 + k) * 128 + e]);
  }
}

// ---------------- input instance-norm stats --------------------------------
__global__ __launch_bounds__(256) void stats_kernel(const float* __restrict__ x,
                                                    float* __restrict__ means,
                                                    float* __restrict__ stdev) {
  int wave = threadIdx.x >> 5, lane = threadIdx.x & 31;
  int wid = blockIdx.x * 8 + wave;       // 512 waves: (b, c)
  int b = wid >> 6, c = wid & 63;
  float s1 = 0.f, s2 = 0.f;
  for (int s = lane; s < Ss; s += 32) {
    float v = x[((size_t)(b << 10) + s) * CIN + c];
    s1 += v; s2 += v * v;
  }
#pragma unroll
  for (int m = 16; m >= 1; m >>= 1) { s1 += __shfl_xor(s1, m, 32); s2 += __shfl_xor(s2, m, 32); }
  if (lane == 0) {
    float mu = s1 * (1.f / Ss);
    float var = s2 * (1.f / Ss) - mu * mu;
    means[wid] = mu;
    stdev[wid] = sqrtf(var + 1e-5f);
  }
}

__global__ void embed_norm_kernel(const float* __restrict__ x, const float* __restrict__ means,
                                  const float* __restrict__ stdev, __bf16* __restrict__ out, int n) {
  for (int i = blockIdx.x * blockDim.x + threadIdx.x; i < n; i += gridDim.x * blockDim.x) {
    int c = i & 63, row = i >> 6, b = row >> 10;
    out[i] = f2bf((x[i] - means[b * 64 + c]) / stdev[b * 64 + c]);
  }
}

// ---------------- generic bf16 WMMA GEMM -----------------------------------
// C(M,N) = A(M,K)@B(K,N) [+bias per col][+resid], f32 out.  128x128 block,
// 8 waves each computing a 32x64 patch of 16x16 WMMA tiles; BK=32 LDS staging.
__global__ __launch_bounds__(256) void gemm_bf16_kernel(
    const __bf16* __restrict__ A, int lda,
    const __bf16* __restrict__ Bm, int ldb,
    float* __restrict__ C, int ldc,
    const float* __restrict__ bias,
    const float* __restrict__ resid, int ldr,
    int M, int N, int K) {
  __shared__ __attribute__((aligned(16))) __bf16 As[128][32];
  __shared__ __attribute__((aligned(16))) __bf16 Bs[128][32];   // transposed [n][k]
  int tid = threadIdx.x, wave = tid >> 5, lane = tid & 31;
  int col = lane & 15, half = lane >> 4;
  int bm = blockIdx.y * 128, bn = blockIdx.x * 128;
  int wm = (wave & 3) * 32, wn = (wave >> 2) * 64;
  v8f vz = {0.f, 0.f, 0.f, 0.f, 0.f, 0.f, 0.f, 0.f};
  v8f acc[2][4];
#pragma unroll
  for (int mi = 0; mi < 2; ++mi)
#pragma unroll
    for (int ni = 0; ni < 4; ++ni) acc[mi][ni] = vz;

  for (int k0 = 0; k0 < K; k0 += 32) {
    for (int i = tid; i < 2048; i += 256) {            // A tile as u32 pairs
      int r = i >> 4, c2 = (i & 15) * 2;
      int gr = bm + r, gc = k0 + c2;
      unsigned val = 0;
      if (gr < M && gc < K) val = *(const unsigned*)(A + (size_t)gr * lda + gc);
      *(unsigned*)(&As[r][c2]) = val;
    }
    for (int i = tid; i < 4096; i += 256) {            // B tile, transpose in LDS
      int kk = i >> 7, n = i & 127;
      int gk = k0 + kk, gn = bn + n;
      __bf16 v = f2bf(0.f);
      if (gk < K && gn < N) v = Bm[(size_t)gk * ldb + gn];
      Bs[n][kk] = v;
    }
    if (k0 + 32 < K && tid < 128 && (bm + tid) < M)    // pull next A tile toward L2
      __builtin_prefetch(A + (size_t)(bm + tid) * lda + k0 + 32, 0, 0);
    __syncthreads();
    v16bf af[2], bfr[4];
#pragma unroll
    for (int mi = 0; mi < 2; ++mi) af[mi] = frag_load_pairs(&As[wm + mi * 16][0], 32, lane);
#pragma unroll
    for (int ni = 0; ni < 4; ++ni) bfr[ni] = frag_load_pairs(&Bs[wn + ni * 16][0], 32, lane);
#pragma unroll
    for (int mi = 0; mi < 2; ++mi)
#pragma unroll
      for (int ni = 0; ni < 4; ++ni) acc[mi][ni] = WMMA_BF16(af[mi], bfr[ni], acc[mi][ni]);
    __syncthreads();
  }
#pragma unroll
  for (int mi = 0; mi < 2; ++mi)
#pragma unroll
    for (int ni = 0; ni < 4; ++ni) {
      int gcol = bn + wn + ni * 16 + col;
      if (gcol >= N) continue;
      float bv = bias ? bias[gcol] : 0.f;
#pragma unroll
      for (int r = 0; r < 8; ++r) {
        int grow = bm + wm + mi * 16 + r + 8 * half;
        if (grow < M) {
          float v = acc[mi][ni][r] + bv;
          if (resid) v += resid[(size_t)grow * ldr + gcol];
          C[(size_t)grow * ldc + gcol] = v;
        }
      }
    }
}

// ---------------- LayerNorm (D=512), bf16 (+optional f32) out --------------
__global__ __launch_bounds__(256) void ln_kernel(const float* __restrict__ X,
                                                 const float* __restrict__ w,
                                                 __bf16* __restrict__ outb,
                                                 float* __restrict__ outf) {
  __shared__ float sb[8];
  int row = blockIdx.x;
  const float* xr = X + (size_t)row * Dd;
  float v0 = xr[threadIdx.x], v1 = xr[threadIdx.x + 256];
  float s1 = block_reduce_256(v0 + v1, sb);
  float s2 = block_reduce_256(v0 * v0 + v1 * v1, sb);
  float mu = s1 * (1.f / Dd);
  float rs = rsqrtf(s2 * (1.f / Dd) - mu * mu + 1e-5f);
#pragma unroll
  for (int j = 0; j < 2; ++j) {
    int d = threadIdx.x + j * 256;
    float y = (xr[d] - mu) * rs * w[d];
    outb[(size_t)row * Dd + d] = f2bf(y);
    if (outf) outf[(size_t)row * Dd + d] = y;
  }
}

// ---------------- depthwise causal conv (K=4) + SiLU -----------------------
__global__ void conv_silu_kernel(const float* __restrict__ xin, int ldin,
                                 const float* __restrict__ w, const float* __restrict__ bias,
                                 int Cc, float* __restrict__ outf, int ldof,
                                 __bf16* __restrict__ outb, int ldob, int total) {
  for (int id = blockIdx.x * blockDim.x + threadIdx.x; id < total; id += gridDim.x * blockDim.x) {
    int c = id % Cc, row = id / Cc;
    int s = row & 1023;
    float a = bias[c];
#pragma unroll
    for (int k = 0; k < 4; ++k) {
      int sp = s - 3 + k;
      if (sp >= 0) a += xin[(size_t)(row - s + sp) * ldin + c] * w[c * 4 + k];
    }
    float y = a / (1.f + __expf(-a));
    outf[(size_t)row * ldof + c] = y;
    if (outb) outb[(size_t)row * ldob + c] = f2bf(y);
  }
}

// ---------------- block-diagonal 4x4 q/k/v projections ---------------------
__global__ void blockwise_qkv_kernel(const float* __restrict__ XCVp,
                                     const float* __restrict__ XMp, int ldxm,
                                     const float* __restrict__ qW, const float* __restrict__ kW,
                                     const float* __restrict__ vW,
                                     __bf16* __restrict__ Qb, __bf16* __restrict__ Kb,
                                     __bf16* __restrict__ Vb, int total) {
  for (int id = blockIdx.x * blockDim.x + threadIdx.x; id < total; id += gridDim.x * blockDim.x) {
    int c = id & 1023, row = id >> 10;
    int nb = c >> 2, e = c & 3;
    float q = 0.f, kv = 0.f, vv = 0.f;
#pragma unroll
    for (int dd = 0; dd < 4; ++dd) {
      float xc = XCVp[(size_t)row * 1024 + nb * 4 + dd];
      float xm = XMp[(size_t)row * ldxm + nb * 4 + dd];
      int wi = (nb * 4 + dd) * 4 + e;
      q += xc * qW[wi]; kv += xc * kW[wi]; vv += xm * vW[wi];
    }
    size_t o = (size_t)row * 1024 + c;
    Qb[o] = f2bf(q); Kb[o] = f2bf(kv); Vb[o] = f2bf(vv);
  }
}

// ---------------- ig/fg gate projections (GEMV over 3072) ------------------
__global__ __launch_bounds__(256) void gates_kernel(
    const __bf16* __restrict__ Qb, const __bf16* __restrict__ Kb, const __bf16* __restrict__ Vb,
    const float* __restrict__ igW, const float* __restrict__ igb,
    const float* __restrict__ fgW, const float* __restrict__ fgb,
    float* __restrict__ IG, float* __restrict__ FG) {
  int wave = threadIdx.x >> 5, lane = threadIdx.x & 31;
  int row = blockIdx.x * 8 + wave;                 // 8192 rows
  int b = row >> 10, s = row & 1023;
  float ai[4] = {0.f, 0.f, 0.f, 0.f}, af[4] = {0.f, 0.f, 0.f, 0.f};
  const __bf16* bufs[3] = {Qb + (size_t)row * 1024, Kb + (size_t)row * 1024, Vb + (size_t)row * 1024};
  for (int t3 = 0; t3 < 3; ++t3) {
    const __bf16* p = bufs[t3];
    for (int j = lane; j < 1024; j += 32) {
      float x = bf2f(p[j]);
      int idx = t3 * 1024 + j;
#pragma unroll
      for (int g = 0; g < 4; ++g) { ai[g] += x * igW[idx * 4 + g]; af[g] += x * fgW[idx * 4 + g]; }
    }
  }
#pragma unroll
  for (int g = 0; g < 4; ++g) {
#pragma unroll
    for (int m = 16; m >= 1; m >>= 1) { ai[g] += __shfl_xor(ai[g], m, 32); af[g] += __shfl_xor(af[g], m, 32); }
  }
  if (lane == 0) {
#pragma unroll
    for (int g = 0; g < 4; ++g) {
      IG[(((size_t)b * 4 + g) << 10) + s] = ai[g] + igb[g];
      FG[(((size_t)b * 4 + g) << 10) + s] = af[g] + fgb[g];
    }
  }
}

__global__ void cumsum_kernel(const float* __restrict__ FG, float* __restrict__ CF) {
  int id = threadIdx.x;                            // 32 threads = (b,h)
  const float* f = FG + (size_t)id * Ss;
  float* c = CF + (size_t)id * Ss;
  float a = 0.f;
  for (int s = 0; s < Ss; ++s) {
    float x = f[s];
    a += fminf(x, 0.f) - log1pf(__expf(-fabsf(x)));
    c[s] = a;
  }
}

// ---------------- mLSTM gated attention (flash-style, WMMA) ----------------
// wave = (b, head, 16-query tile, d-half of 128).  Online rescale by running
// row-max of logD; final norm = max(|rowsum|, exp(-max)) + 1e-6.
__global__ __launch_bounds__(256) void mlstm_attn_kernel(
    const __bf16* __restrict__ Qb, const __bf16* __restrict__ Kb, const __bf16* __restrict__ Vb,
    const float* __restrict__ cf, const float* __restrict__ ig, float* __restrict__ Hout) {
  __shared__ __attribute__((aligned(16))) __bf16 Pl[8][16][32];
  int wave = threadIdx.x >> 5, lane = threadIdx.x & 31;
  int col = lane & 15, half = lane >> 4;
  int wid = blockIdx.x * 8 + wave;
  int dhalf = wid & 1; wid >>= 1;
  int tt = wid & 63; wid >>= 6;
  int h = wid & 3; int b = wid >> 2;

#pragma unroll
  for (int r = 0; r < 8; ++r) Pl[wave][r + 8 * half][16 + col] = f2bf(0.f);  // zero K>=16

  const __bf16* qbase = Qb + ((size_t)(b * Ss) + tt * 16) * 1024 + h * 256;
  v16bf qf[8];
#pragma unroll
  for (int kk = 0; kk < 8; ++kk) qf[kk] = frag_load_pairs(qbase + kk * 32, 1024, lane);

  const float* cfbh = cf + ((size_t)b * 4 + h) * Ss;
  const float* igbh = ig + ((size_t)b * 4 + h) * Ss;
  float cfT[8];
#pragma unroll
  for (int r = 0; r < 8; ++r) cfT[r] = cfbh[tt * 16 + r + 8 * half];

  v8f vz = {0.f, 0.f, 0.f, 0.f, 0.f, 0.f, 0.f, 0.f};
  float mrun[8], lsum[8];
  v8f oacc[8];
#pragma unroll
  for (int r = 0; r < 8; ++r) { mrun[r] = -3.0e38f; lsum[r] = 0.f; }
#pragma unroll
  for (int ni = 0; ni < 8; ++ni) oacc[ni] = vz;

  for (int st = 0; st <= tt; ++st) {
    v8f sacc = vz;
    const __bf16* kbase = Kb + ((size_t)(b * Ss) + st * 16) * 1024 + h * 256;
#pragma unroll
    for (int kk = 0; kk < 8; ++kk) {
      v16bf kf = frag_load_pairs(kbase + kk * 32, 1024, lane);
      sacc = WMMA_BF16(qf[kk], kf, sacc);
    }
    float cfS = cfbh[st * 16 + col];
    float igS = igbh[st * 16 + col];
    float scale[8];
#pragma unroll
    for (int r = 0; r < 8; ++r) {
      int t = tt * 16 + r + 8 * half, s = st * 16 + col;
      bool valid = (t >= s);
      float logd = valid ? (cfT[r] - cfS + igS) : -3.0e38f;
      float mx = logd;
#pragma unroll
      for (int m = 1; m < 16; m <<= 1) mx = fmaxf(mx, __shfl_xor(mx, m, 32));
      float mnew = fmaxf(mrun[r], mx);
      float sc = __expf(mrun[r] - mnew);
      float p = valid ? (sacc[r] * 0.0625f) * __expf(logd - mnew) : 0.f;
      float ps = p;
#pragma unroll
      for (int m = 1; m < 16; m <<= 1) ps += __shfl_xor(ps, m, 32);
      lsum[r] = lsum[r] * sc + ps;
      mrun[r] = mnew;
      scale[r] = sc;
      Pl[wave][r + 8 * half][col] = f2bf(p);
    }
#pragma unroll
    for (int ni = 0; ni < 8; ++ni)
#pragma unroll
      for (int r = 0; r < 8; ++r) oacc[ni][r] *= scale[r];
    asm volatile("s_wait_dscnt 0" ::: "memory");          // LDS P tile visible to frag read
    v16bf pf = frag_load_pairs(&Pl[wave][0][0], 32, lane);
    const __bf16* vbase = Vb + ((size_t)(b * Ss) + st * 16) * 1024 + h * 256 + dhalf * 128;
#pragma unroll
    for (int ni = 0; ni < 8; ++ni) {
      v16bf vf = frag_load_kmajor(vbase + ni * 16, 1024, lane, 16);
      oacc[ni] = WMMA_BF16(pf, vf, oacc[ni]);
    }
  }
#pragma unroll
  for (int r = 0; r < 8; ++r) {
    float inv = 1.f / (fmaxf(fabsf(lsum[r]), __expf(-mrun[r])) + 1e-6f);
    int t = tt * 16 + r + 8 * half;
    float* out = Hout + ((size_t)(b * Ss) + t) * 1024 + h * 256 + dhalf * 128;
#pragma unroll
    for (int ni = 0; ni < 8; ++ni) out[ni * 16 + col] = oacc[ni][r] * inv;
  }
}

// ---------------- mLSTM epilogue: head-GN + skip + SiLU(z) gate ------------
__global__ __launch_bounds__(256) void mh_fuse_kernel(
    const float* __restrict__ HATT, const float* __restrict__ UP,
    const float* __restrict__ XCV, const float* __restrict__ mhw,
    const float* __restrict__ skip, __bf16* __restrict__ HN) {
  int wave = threadIdx.x >> 5, lane = threadIdx.x & 31;
  int wid = blockIdx.x * 8 + wave;
  int h = wid & 3, row = wid >> 2;
  const float* hp = HATT + (size_t)row * 1024 + h * 256;
  float v[8], s1 = 0.f, s2 = 0.f;
#pragma unroll
  for (int j = 0; j < 8; ++j) { v[j] = hp[j * 32 + lane]; s1 += v[j]; s2 += v[j] * v[j]; }
#pragma unroll
  for (int m = 16; m >= 1; m >>= 1) { s1 += __shfl_xor(s1, m, 32); s2 += __shfl_xor(s2, m, 32); }
  float mu = s1 * (1.f / 256.f);
  float rs = rsqrtf(s2 * (1.f / 256.f) - mu * mu + 1e-5f);
#pragma unroll
  for (int j = 0; j < 8; ++j) {
    int d = j * 32 + lane, c = h * 256 + d;
    float z = UP[(size_t)row * 2048 + 1024 + c];
    float g = z / (1.f + __expf(-z));
    float o = ((v[j] - mu) * rs * mhw[c] + skip[c] * XCV[(size_t)row * 1024 + c]) * g;
    HN[(size_t)row * 1024 + c] = f2bf(o);
  }
}

// ---------------- sLSTM persistent scan (1 WG per head) --------------------
__global__ __launch_bounds__(256) void slstm_scan_kernel(
    const float* __restrict__ gx, const __bf16* __restrict__ Rp, float* __restrict__ hs) {
  __shared__ __attribute__((aligned(16))) __bf16 hlds[16][128];
  int h = blockIdx.x;
  int tid = threadIdx.x, wave = tid >> 5, lane = tid & 31;
  int col = lane & 15, half = lane >> 4;
  for (int i = tid; i < 16 * 128; i += 256) ((__bf16*)hlds)[i] = f2bf(0.f);
  float cs[8], ns[8], ms[8];
#pragma unroll
  for (int r = 0; r < 8; ++r) { cs[r] = 0.f; ns[r] = 0.f; ms[r] = 0.f; }
  const __bf16* Rh = Rp + (size_t)h * 128 * 512;   // [k=128][n=512], L2-resident
  v8f vz = {0.f, 0.f, 0.f, 0.f, 0.f, 0.f, 0.f, 0.f};
  __syncthreads();
  for (int t = 0; t < Ss; ++t) {
    v8f acc[4] = {vz, vz, vz, vz};
#pragma unroll
    for (int kk = 0; kk < 4; ++kk) {
      v16bf a = frag_load_pairs(&hlds[0][kk * 32], 128, lane);
#pragma unroll
      for (int g = 0; g < 4; ++g) {
        int nb = g * 128 + wave * 16;
        v16bf bfr = frag_load_kmajor(Rh + (size_t)(kk * 32) * 512 + nb, 512, lane, 32);
        acc[g] = WMMA_BF16(a, bfr, acc[g]);
      }
    }
    __syncthreads();                               // all hlds reads done
    if (half == 0) {                               // rows 0..7 = batches
      int e = wave * 16 + col;
#pragma unroll
      for (int r = 0; r < 8; ++r) {
        size_t gbase = ((size_t)r * Ss + t) * 2048 + (size_t)h * 512;
        float it = acc[0][r] + gx[gbase + 0 * 128 + e];
        float ft = acc[1][r] + gx[gbase + 1 * 128 + e];
        float zt = acc[2][r] + gx[gbase + 2 * 128 + e];
        float ot = acc[3][r] + gx[gbase + 3 * 128 + e];
        float mn = fmaxf(ft + ms[r], it);
        float iv = __expf(it - mn), fv = __expf(ft + ms[r] - mn);
        float c = fv * cs[r] + iv * tanhf(zt);
        float n = fv * ns[r] + iv;
        float hv = (1.f / (1.f + __expf(-ot))) * c / n;
        cs[r] = c; ns[r] = n; ms[r] = mn;
        hlds[r][e] = f2bf(hv);
        hs[((size_t)r * Ss + t) * 512 + h * 128 + e] = hv;
      }
    }
    __syncthreads();                               // h(t) visible for step t+1
  }
}

// ---------------- sLSTM GroupNorm + residual -------------------------------
__global__ __launch_bounds__(256) void slstm_gn_res_kernel(const float* __restrict__ HS,
                                                           const float* __restrict__ gnw,
                                                           float* __restrict__ X) {
  int wave = threadIdx.x >> 5, lane = threadIdx.x & 31;
  int wid = blockIdx.x * 8 + wave;
  int h = wid & 3, row = wid >> 2;
  const float* hp = HS + (size_t)row * 512 + h * 128;
  float v[4], s1 = 0.f, s2 = 0.f;
#pragma unroll
  for (int j = 0; j < 4; ++j) { v[j] = hp[j * 32 + lane]; s1 += v[j]; s2 += v[j] * v[j]; }
#pragma unroll
  for (int m = 16; m >= 1; m >>= 1) { s1 += __shfl_xor(s1, m, 32); s2 += __shfl_xor(s2, m, 32); }
  float mu = s1 * (1.f / 128.f);
  float rs = rsqrtf(s2 * (1.f / 128.f) - mu * mu + 1e-5f);
#pragma unroll
  for (int j = 0; j < 4; ++j) {
    int d = j * 32 + lane;
    X[(size_t)row * 512 + h * 128 + d] += (v[j] - mu) * rs * gnw[h * 128 + d];
  }
}

// ---------------- GEGLU activation -----------------------------------------
__global__ void geglu_kernel(const float* __restrict__ U, int ldu,
                             __bf16* __restrict__ out, int ldo, int total) {
  for (int id = blockIdx.x * blockDim.x + threadIdx.x; id < total; id += gridDim.x * blockDim.x) {
    int c = id % FFUP, row = id / FFUP;
    float g = U[(size_t)row * ldu + c], u = U[(size_t)row * ldu + FFUP + c];
    float tt = 0.7978845608f * (g + 0.044715f * g * g * g);
    float gl = 0.5f * g * (1.f + tanhf(tt));
    out[(size_t)row * ldo + c] = f2bf(gl * u);
  }
}

// ---------------- final LN + projection + de-normalization -----------------
__global__ __launch_bounds__(256) void final_kernel(const float* __restrict__ X,
                                                    const float* __restrict__ postw,
                                                    const float* __restrict__ projW,
                                                    const float* __restrict__ projb,
                                                    const float* __restrict__ means,
                                                    const float* __restrict__ stdev,
                                                    float* __restrict__ out) {
  __shared__ float xs[512];
  __shared__ float sb[8];
  int b = blockIdx.x;
  const float* xr = X + ((size_t)b * Ss + (Ss - 1)) * Dd;
  float v0 = xr[threadIdx.x], v1 = xr[threadIdx.x + 256];
  float s1 = block_reduce_256(v0 + v1, sb);
  float s2 = block_reduce_256(v0 * v0 + v1 * v1, sb);
  float mu = s1 * (1.f / Dd);
  float rs = rsqrtf(s2 * (1.f / Dd) - mu * mu + 1e-5f);
#pragma unroll
  for (int j = 0; j < 2; ++j) {
    int d = threadIdx.x + j * 256;
    xs[d] = (xr[d] - mu) * rs * postw[d];
  }
  __syncthreads();
  for (int c = threadIdx.x; c < 64; c += 256) {
    float acc = projb[c];
    for (int k = 0; k < Dd; ++k) acc += xs[k] * projW[k * 64 + c];
    out[b * 64 + c] = acc * stdev[b * 64 + c] + means[b * 64 + c];
  }
}

// ===========================================================================
extern "C" void kernel_launch(void* const* d_in, const int* in_sizes, int n_in,
                              void* d_out, int out_size, void* d_ws, size_t ws_size,
                              hipStream_t stream) {
  (void)in_sizes; (void)n_in; (void)out_size; (void)ws_size;
  // params flattened in sorted pytree order after x_enc:
  const float* x_enc   = (const float*)d_in[0];
  const float* b0convW = (const float*)d_in[1];
  const float* b0convB = (const float*)d_in[2];
  const float* b0downW = (const float*)d_in[3];
  const float* b0fgW   = (const float*)d_in[4];
  const float* b0fgB   = (const float*)d_in[5];
  const float* b0igW   = (const float*)d_in[6];
  const float* b0igB   = (const float*)d_in[7];
  const float* b0kW    = (const float*)d_in[8];
  const float* b0lnW   = (const float*)d_in[9];
  const float* b0mhW   = (const float*)d_in[10];
  const float* b0qW    = (const float*)d_in[11];
  const float* b0skip  = (const float*)d_in[12];
  const float* b0upW   = (const float*)d_in[13];
  const float* b0vW    = (const float*)d_in[14];
  const float* b1R     = (const float*)d_in[15];
  const float* b1Wg    = (const float*)d_in[16];
  const float* b1bg    = (const float*)d_in[17];
  const float* b1convW = (const float*)d_in[18];
  const float* b1convB = (const float*)d_in[19];
  const float* b1ffdW  = (const float*)d_in[20];
  const float* b1ffuW  = (const float*)d_in[21];
  const float* b1gnW   = (const float*)d_in[22];
  const float* b1ln2W  = (const float*)d_in[23];
  const float* b1lnW   = (const float*)d_in[24];
  const float* embW    = (const float*)d_in[25];
  const float* embB    = (const float*)d_in[26];
  const float* postW   = (const float*)d_in[27];
  const float* projW   = (const float*)d_in[28];
  const float* projB   = (const float*)d_in[29];
  float* out = (float*)d_out;

  // ---- workspace layout (~220 MB) ----
  char* ws = (char*)d_ws;
  size_t o = 0;
  auto alloc = [&](size_t bytes) -> char* {
    o = (o + 255) & ~(size_t)255;
    char* p = ws + o; o += bytes; return p;
  };
  float*  MEANS = (float*)alloc(512 * 4);
  float*  STDEV = (float*)alloc(512 * 4);
  float*  CF    = (float*)alloc(32 * Ss * 4);
  float*  IG    = (float*)alloc(32 * Ss * 4);
  float*  FG    = (float*)alloc(32 * Ss * 4);
  __bf16* XN    = (__bf16*)alloc((size_t)ROWS * 64 * 2);
  float*  X     = (float*)alloc((size_t)ROWS * 512 * 4);
  __bf16* LNB   = (__bf16*)alloc((size_t)ROWS * 512 * 2);
  float*  UP    = (float*)alloc((size_t)ROWS * 2048 * 4);   // xm|z ; gx ; ffn-up
  float*  XCV   = (float*)alloc((size_t)ROWS * 1024 * 4);   // b0 conv ; b1 conv
  float*  HATT  = (float*)alloc((size_t)ROWS * 1024 * 4);   // attn out ; b1 ln f32 ; hs
  __bf16* QB    = (__bf16*)alloc((size_t)ROWS * 1024 * 2);  // q ; HN
  __bf16* KB    = (__bf16*)alloc((size_t)ROWS * 1024 * 2);  // k ; xc_bf ; ff act
  __bf16* VB    = (__bf16*)alloc((size_t)ROWS * 1024 * 2);  // v
  __bf16* EMBW  = (__bf16*)alloc(64 * 512 * 2);
  __bf16* UPW   = (__bf16*)alloc((size_t)512 * 2048 * 2);
  __bf16* DOWNW = (__bf16*)alloc((size_t)1024 * 512 * 2);
  __bf16* WGP   = (__bf16*)alloc((size_t)4 * 128 * 512 * 2);
  __bf16* RP    = (__bf16*)alloc((size_t)4 * 128 * 512 * 2);
  __bf16* FFUW  = (__bf16*)alloc((size_t)512 * 1408 * 2);
  __bf16* FFDW  = (__bf16*)alloc((size_t)704 * 512 * 2);

  auto gemm = [&](const __bf16* A, int lda, const __bf16* Bm, int ldb,
                  float* C, int ldc, const float* bias, const float* resid, int ldr,
                  int M, int N, int K) {
    dim3 g((N + 127) / 128, (M + 127) / 128);
    gemm_bf16_kernel<<<g, 256, 0, stream>>>(A, lda, Bm, ldb, C, ldc, bias, resid, ldr, M, N, K);
  };
  auto cvt = [&](const float* s, __bf16* d, int n) {
    cvt_f32_bf16_kernel<<<dim3((n + 255) / 256 > 2048 ? 2048 : (n + 255) / 256), 256, 0, stream>>>(s, d, n);
  };

  // ---- weight prep (bf16) ----
  cvt(embW,   EMBW,  64 * 512);
  cvt(b0upW,  UPW,   512 * 2048);
  cvt(b0downW,DOWNW, 1024 * 512);
  cvt(b1R,    RP,    4 * 128 * 512);
  cvt(b1ffuW, FFUW,  512 * 1408);
  cvt(b1ffdW, FFDW,  704 * 512);
  pack_wg_kernel<<<dim3(512), 256, 0, stream>>>(b1Wg, WGP);

  // ---- input instance-norm + embed ----
  stats_kernel<<<dim3(64), 256, 0, stream>>>(x_enc, MEANS, STDEV);
  embed_norm_kernel<<<dim3(2048), 256, 0, stream>>>(x_enc, MEANS, STDEV, XN, ROWS * 64);
  gemm(XN, 64, EMBW, 512, X, 512, embB, nullptr, 0, ROWS, 512, 64);

  // ---- mLSTM block ----
  ln_kernel<<<dim3(ROWS), 256, 0, stream>>>(X, b0lnW, LNB, nullptr);
  gemm(LNB, 512, UPW, 2048, UP, 2048, nullptr, nullptr, 0, ROWS, 2048, 512);      // xm|z
  conv_silu_kernel<<<dim3(4096), 256, 0, stream>>>(UP, 2048, b0convW, b0convB, 1024,
                                                   XCV, 1024, nullptr, 0, ROWS * 1024);
  blockwise_qkv_kernel<<<dim3(4096), 256, 0, stream>>>(XCV, UP, 2048, b0qW, b0kW, b0vW,
                                                       QB, KB, VB, ROWS * 1024);
  gates_kernel<<<dim3(1024), 256, 0, stream>>>(QB, KB, VB, b0igW, b0igB, b0fgW, b0fgB, IG, FG);
  cumsum_kernel<<<dim3(1), 32, 0, stream>>>(FG, CF);
  mlstm_attn_kernel<<<dim3(512), 256, 0, stream>>>(QB, KB, VB, CF, IG, HATT);
  mh_fuse_kernel<<<dim3(4096), 256, 0, stream>>>(HATT, UP, XCV, b0mhW, b0skip, QB);  // QB := HN
  gemm(QB, 1024, DOWNW, 512, X, 512, nullptr, X, 512, ROWS, 512, 1024);              // x += down

  // ---- sLSTM block ----
  ln_kernel<<<dim3(ROWS), 256, 0, stream>>>(X, b1lnW, LNB, HATT);                    // bf16 + f32
  conv_silu_kernel<<<dim3(4096), 256, 0, stream>>>(HATT, 512, b1convW, b1convB, 512,
                                                   XCV, 512, KB, 512, ROWS * 512);   // KB := xc_bf
  for (int h = 0; h < 4; ++h) {
    const __bf16* Wh = WGP + (size_t)h * 128 * 512;
    // gates 0,1 from conv(xc); gates 2,3 from ln(x)
    gemm(KB  + h * 128, 512, Wh,       512, UP + h * 512,       2048,
         b1bg + h * 512,       nullptr, 0, ROWS, 256, 128);
    gemm(LNB + h * 128, 512, Wh + 256, 512, UP + h * 512 + 256, 2048,
         b1bg + h * 512 + 256, nullptr, 0, ROWS, 256, 128);
  }
  slstm_scan_kernel<<<dim3(4), 256, 0, stream>>>(UP, RP, HATT);                      // HATT := hs
  slstm_gn_res_kernel<<<dim3(4096), 256, 0, stream>>>(HATT, b1gnW, X);

  // ---- FFN ----
  ln_kernel<<<dim3(ROWS), 256, 0, stream>>>(X, b1ln2W, LNB, nullptr);
  gemm(LNB, 512, FFUW, 1408, UP, 1408, nullptr, nullptr, 0, ROWS, 1408, 512);
  geglu_kernel<<<dim3(4096), 256, 0, stream>>>(UP, 1408, KB, 704, ROWS * FFUP);      // KB := act
  gemm(KB, 704, FFDW, 512, X, 512, nullptr, X, 512, ROWS, 512, 704);

  // ---- head ----
  final_kernel<<<dim3(8), 256, 0, stream>>>(X, postW, projW, projB, MEANS, STDEV, out);
}
